// Net_90074054132252
// MI455X (gfx1250) — matrix operand order repaired and checked by reading, not compiled, not run on hardware
//
#include <hip/hip_runtime.h>
#include <hip/hip_bf16.h>

// ---------------------------------------------------------------------------
// GAT autoencoder forward for MI455X (gfx1250).
// GEMMs use v_wmma_f32_16x16x32_f16 (wave32 WMMA). Attention uses fp32
// atomics resolved in L2 (whole per-layer working set fits in 192MB L2).
// ---------------------------------------------------------------------------

#define N_NODES 50000
#define N_EDGES 800000
#define E2      (N_EDGES + N_NODES)   // with self loops
#define F_IN    42
#define HID     128

typedef __attribute__((ext_vector_type(16))) _Float16 v16h;
typedef __attribute__((ext_vector_type(8)))  _Float16 v8h;
typedef __attribute__((ext_vector_type(8)))  float    v8f;

static inline int cdiv(long long a, long long b) { return (int)((a + b - 1) / b); }

// ----------------------------- small helpers -------------------------------

__device__ inline void atomicMaxF(float* addr, float v) {
    // float max via monotone int/uint mapping (valid with -inf init)
    if (v >= 0.0f) atomicMax((int*)addr, __float_as_int(v));
    else           atomicMin((unsigned int*)addr, __float_as_uint(v));
}

// ----------------------------- kernels -------------------------------------

// src/dst with appended self loops
__global__ void build_edges(const int* __restrict__ ei, int* __restrict__ srcv,
                            int* __restrict__ dstv) {
    int i = blockIdx.x * blockDim.x + threadIdx.x;
    if (i >= E2) return;
    if (i < N_EDGES) { srcv[i] = ei[i]; dstv[i] = ei[N_EDGES + i]; }
    else             { srcv[i] = i - N_EDGES; dstv[i] = i - N_EDGES; }
}

// xn = x / max(sum|x|,1e-12), written fp16 padded to 64 cols
__global__ void normalize_x(const float* __restrict__ x, _Float16* __restrict__ xh) {
    int n = blockIdx.x * blockDim.x + threadIdx.x;
    if (n >= N_NODES) return;
    const float* xr = x + (size_t)n * F_IN;
    float s = 0.f;
    for (int k = 0; k < F_IN; ++k) s += fabsf(xr[k]);
    float inv = 1.0f / fmaxf(s, 1e-12f);
    _Float16* o = xh + (size_t)n * 64;
    for (int k = 0; k < 64; ++k)
        o[k] = (k < F_IN) ? (_Float16)(xr[k] * inv) : (_Float16)0.0f;
}

// Wt[o][k] = (half)W[k][o], zero padded to KinPad x KoutPad
__global__ void convert_wt(const float* __restrict__ W, _Float16* __restrict__ Wt,
                           int Kin, int Kout, int KinPad, int KoutPad) {
    int i = blockIdx.x * blockDim.x + threadIdx.x;
    if (i >= KoutPad * KinPad) return;
    int o = i / KinPad, k = i - o * KinPad;
    float v = (o < Kout && k < Kin) ? W[(size_t)k * Kout + o] : 0.0f;
    Wt[i] = (_Float16)v;
}

// H[N,Kout] = Xh[N,Kpad] @ Wt^T ; one wave per 16x16 output tile, K step 32.
// Lane layouts: A and B both use contiguous k = 16*hi + j per lane (not the ISA table order).
__global__ __launch_bounds__(256)
void gemm_wmma_f16(const _Float16* __restrict__ Xh, const _Float16* __restrict__ Wt,
                   float* __restrict__ H, int Kpad, int Kout, int KoutPad) {
    const int tilesN = KoutPad >> 4;
    const int totalWaves = (N_NODES >> 4) * tilesN;
    const int gw = blockIdx.x * (blockDim.x >> 5) + (threadIdx.x >> 5);
    if (gw >= totalWaves) return;                 // wave-uniform exit (EXEC stays all-1)
    const int tileM = gw / tilesN;
    const int tileN = gw - tileM * tilesN;
    const int lane  = threadIdx.x & 31;
    const int lid   = lane & 15;
    const int hi    = lane >> 4;

    const _Float16* arow = Xh + (size_t)(tileM * 16 + lid) * Kpad;
    const _Float16* brow = Wt + (size_t)(tileN * 16 + lid) * Kpad;

    v8f c = {};
    for (int k0 = 0; k0 < Kpad; k0 += 32) {
        // A: element j -> k = k0 + j + 16*hi  => two contiguous b128 loads
        v8h alo = *(const v8h*)(arow + k0 + hi * 16);
        v8h ahi = *(const v8h*)(arow + k0 + hi * 16 + 8);
        v16h a;
#pragma unroll
        for (int t = 0; t < 8; ++t) { a[t] = alo[t]; a[t + 8] = ahi[t]; }
        // B: element j -> k = k0 + j + 16*hi => one contiguous 32B load
        v16h b = *(const v16h*)(brow + k0 + hi * 16);
        c = __builtin_amdgcn_wmma_f32_16x16x32_f16(false, a, false, b,
                                                   (short)0, c, false, false);
    }
    const int outn = tileN * 16 + lid;
    if (outn < Kout) {
#pragma unroll
        for (int r = 0; r < 8; ++r) {             // D: row = r + 8*hi, col = lid
            const int mr = tileM * 16 + r + hi * 8;
            H[(size_t)mr * Kout + outn] = c[r];
        }
    }
}

// hs = h.a_s, hd = h.a_d ; also init segment max/den
__global__ void dots_init(const float* __restrict__ h, const float* __restrict__ as,
                          const float* __restrict__ ad, int Kout,
                          float* __restrict__ hs, float* __restrict__ hd,
                          float* __restrict__ mx, float* __restrict__ den) {
    int n = blockIdx.x * blockDim.x + threadIdx.x;
    if (n >= N_NODES) return;
    const float* r = h + (size_t)n * Kout;
    float s = 0.f, d = 0.f;
    for (int k = 0; k < Kout; ++k) { float v = r[k]; s += v * as[k]; d += v * ad[k]; }
    hs[n] = s; hd[n] = d;
    mx[n] = -3.402823466e38f; den[n] = 0.0f;
}

__global__ void zero_f32(float* __restrict__ p, long long cnt) {
    long long i = (long long)blockIdx.x * blockDim.x + threadIdx.x;
    if (i < cnt) p[i] = 0.0f;
}

// pass 1: e = leaky_relu(hs[src]+hd[dst]); segment max into mx[dst]
__global__ void edge_max(const int* __restrict__ srcv, const int* __restrict__ dstv,
                         const float* __restrict__ hs, const float* __restrict__ hd,
                         float* __restrict__ ebuf, float* __restrict__ mx) {
    int i = blockIdx.x * blockDim.x + threadIdx.x;
    if (i >= E2) return;
    float ev = hs[srcv[i]] + hd[dstv[i]];
    ev = ev > 0.0f ? ev : 0.2f * ev;
    ebuf[i] = ev;
    atomicMaxF(&mx[dstv[i]], ev);
}

// pass 2: ex = exp(e - mx[dst]); segment sum into den[dst]
__global__ void edge_exp(const int* __restrict__ dstv, float* __restrict__ ebuf,
                         const float* __restrict__ mx, float* __restrict__ den) {
    int i = blockIdx.x * blockDim.x + threadIdx.x;
    if (i >= E2) return;
    float ex = __expf(ebuf[i] - mx[dstv[i]]);
    ebuf[i] = ex;
    atomicAdd(&den[dstv[i]], ex);
}

// pass 3: out[dst] += h[src] * (ex / (den[dst]+1e-16)) ; one thread per (edge,f)
__global__ void edge_scatter(const int* __restrict__ srcv, const int* __restrict__ dstv,
                             const float* __restrict__ ebuf, const float* __restrict__ den,
                             const float* __restrict__ h, float* __restrict__ out,
                             int Kout) {
    long long tid = (long long)blockIdx.x * blockDim.x + threadIdx.x;
    long long tot = (long long)E2 * Kout;
    if (tid >= tot) return;
    int e = (int)(tid / Kout);
    int f = (int)(tid - (long long)e * Kout);
    int s = srcv[e], d = dstv[e];
    float alpha = ebuf[e] / (den[d] + 1e-16f);
    atomicAdd(&out[(size_t)d * Kout + f], h[(size_t)s * Kout + f] * alpha);
}

// act = tanh(out + b); optional fp32 copy and fp16 copy (input of next GEMM)
__global__ void finalize_act(const float* __restrict__ out, const float* __restrict__ bias,
                             float* __restrict__ actF, _Float16* __restrict__ actH,
                             int Kout) {
    long long i = (long long)blockIdx.x * blockDim.x + threadIdx.x;
    long long tot = (long long)N_NODES * Kout;
    if (i >= tot) return;
    int f = (int)(i % Kout);
    float v = tanhf(out[i] + bias[f]);
    if (actF) actF[i] = v;
    if (actH) actH[i] = (_Float16)v;   // Kout==128 here, already K-aligned
}

// mask MLP 128->64->16->16->2, gate = softmax(m4)[1]; write gated latent (fp16)
__global__ void mask_gate(const float* __restrict__ latent,
                          const float* __restrict__ mw1, const float* __restrict__ mb1,
                          const float* __restrict__ mw2, const float* __restrict__ mb2,
                          const float* __restrict__ mw3, const float* __restrict__ mb3,
                          const float* __restrict__ mw4, const float* __restrict__ mb4,
                          _Float16* __restrict__ actH) {
    int n = blockIdx.x * blockDim.x + threadIdx.x;
    if (n >= N_NODES) return;
    const float* xr = latent + (size_t)n * HID;
    float m1[64];
    for (int o = 0; o < 64; ++o) {
        float acc = mb1[o];
        for (int i = 0; i < HID; ++i) acc += xr[i] * mw1[i * 64 + o];  // uniform -> s_load
        m1[o] = tanhf(acc);
    }
    float m2[16];
    for (int o = 0; o < 16; ++o) {
        float acc = mb2[o];
        for (int i = 0; i < 64; ++i) acc += m1[i] * mw2[i * 16 + o];
        m2[o] = tanhf(acc);
    }
    float m3[16];
    for (int o = 0; o < 16; ++o) {
        float acc = mb3[o];
        for (int i = 0; i < 16; ++i) acc += m2[i] * mw3[i * 16 + o];
        m3[o] = tanhf(acc);
    }
    float z0 = mb4[0], z1 = mb4[1];
    for (int i = 0; i < 16; ++i) { z0 += m3[i] * mw4[i * 2]; z1 += m3[i] * mw4[i * 2 + 1]; }
    float m = fmaxf(z0, z1);
    float e0 = __expf(z0 - m), e1 = __expf(z1 - m);
    float gate = e1 / (e0 + e1);                  // sum(softmax * [0,1]) == p1
    _Float16* o = actH + (size_t)n * HID;
    for (int i = 0; i < HID; ++i) o[i] = (_Float16)(xr[i] * gate);
}

// ----------------------------- host orchestration --------------------------

struct Scratch {
    float *h, *outacc, *actF, *hs, *hd, *mx, *den, *ebuf;
    _Float16 *actH, *Wt;
    int *srcv, *dstv;
};

static void run_gat_layer(const _Float16* Xh, int Kpad,
                          const float* W, const float* as, const float* ad,
                          const float* bias, int Kin, int Kout, int KoutPad,
                          float* actF_out, _Float16* actH_out,
                          const Scratch& S, hipStream_t st) {
    convert_wt<<<cdiv((long long)KoutPad * Kpad, 256), 256, 0, st>>>(W, S.Wt, Kin, Kout, Kpad, KoutPad);
    {
        int totalWaves = (N_NODES / 16) * (KoutPad / 16);
        gemm_wmma_f16<<<cdiv(totalWaves, 8), 256, 0, st>>>(Xh, S.Wt, S.h, Kpad, Kout, KoutPad);
    }
    dots_init<<<cdiv(N_NODES, 256), 256, 0, st>>>(S.h, as, ad, Kout, S.hs, S.hd, S.mx, S.den);
    zero_f32<<<cdiv((long long)N_NODES * Kout, 256), 256, 0, st>>>(S.outacc, (long long)N_NODES * Kout);
    edge_max<<<cdiv(E2, 256), 256, 0, st>>>(S.srcv, S.dstv, S.hs, S.hd, S.ebuf, S.mx);
    edge_exp<<<cdiv(E2, 256), 256, 0, st>>>(S.dstv, S.ebuf, S.mx, S.den);
    edge_scatter<<<cdiv((long long)E2 * Kout, 256), 256, 0, st>>>(S.srcv, S.dstv, S.ebuf, S.den, S.h, S.outacc, Kout);
    finalize_act<<<cdiv((long long)N_NODES * Kout, 256), 256, 0, st>>>(S.outacc, bias, actF_out, actH_out, Kout);
}

extern "C" void kernel_launch(void* const* d_in, const int* in_sizes, int n_in,
                              void* d_out, int out_size, void* d_ws, size_t ws_size,
                              hipStream_t stream) {
    // inputs (setup_inputs order)
    const float* x    = (const float*)d_in[0];
    const int*   ei   = (const int*)d_in[1];
    // d_in[2]=batch, d_in[3]=epoch : unused
    const float* W1   = (const float*)d_in[4];
    const float* a1s  = (const float*)d_in[5];
    const float* a1d  = (const float*)d_in[6];
    const float* b1   = (const float*)d_in[7];
    const float* Wm   = (const float*)d_in[8];    // [6,128,128]
    const float* ams  = (const float*)d_in[9];    // [6,128]
    const float* amd  = (const float*)d_in[10];
    const float* bm   = (const float*)d_in[11];
    const float* W8   = (const float*)d_in[12];   // [128,42]
    const float* a8s  = (const float*)d_in[13];
    const float* a8d  = (const float*)d_in[14];
    const float* b8   = (const float*)d_in[15];
    const float* mw1  = (const float*)d_in[16];
    const float* mb1  = (const float*)d_in[17];
    const float* mw2  = (const float*)d_in[18];
    const float* mb2  = (const float*)d_in[19];
    const float* mw3  = (const float*)d_in[20];
    const float* mb3  = (const float*)d_in[21];
    const float* mw4  = (const float*)d_in[22];
    const float* mb4  = (const float*)d_in[23];
    float* out = (float*)d_out;                   // [N, 42]

    // workspace carve-out (256B aligned slices)
    char* w = (char*)d_ws;
    auto carve = [&](size_t bytes) -> void* {
        void* p = (void*)w;
        w += (bytes + 255) & ~(size_t)255;
        return p;
    };
    Scratch S;
    S.h      = (float*)carve((size_t)N_NODES * HID * 4);
    S.outacc = (float*)carve((size_t)N_NODES * HID * 4);
    S.actF   = (float*)carve((size_t)N_NODES * HID * 4);
    S.actH   = (_Float16*)carve((size_t)N_NODES * HID * 2);
    S.Wt     = (_Float16*)carve((size_t)HID * HID * 2);
    S.hs     = (float*)carve((size_t)N_NODES * 4);
    S.hd     = (float*)carve((size_t)N_NODES * 4);
    S.mx     = (float*)carve((size_t)N_NODES * 4);
    S.den    = (float*)carve((size_t)N_NODES * 4);
    S.ebuf   = (float*)carve((size_t)E2 * 4);
    S.srcv   = (int*)carve((size_t)E2 * 4);
    S.dstv   = (int*)carve((size_t)E2 * 4);

    build_edges<<<cdiv(E2, 256), 256, 0, stream>>>(ei, S.srcv, S.dstv);
    normalize_x<<<cdiv(N_NODES, 256), 256, 0, stream>>>(x, S.actH);   // [N,64] fp16

    // conv1: 42 -> 128 (Kin padded 42->64)
    run_gat_layer(S.actH, 64, W1, a1s, a1d, b1, F_IN, HID, HID,
                  nullptr, S.actH, S, stream);
    // conv2..conv4 (encoder mid); conv4 also emits fp32 latent
    for (int l = 0; l < 3; ++l) {
        run_gat_layer(S.actH, HID, Wm + (size_t)l * HID * HID,
                      ams + (size_t)l * HID, amd + (size_t)l * HID, bm + (size_t)l * HID,
                      HID, HID, HID, (l == 2) ? S.actF : nullptr, S.actH, S, stream);
    }
    // mask MLP gate on latent, write gated fp16 activations
    mask_gate<<<cdiv(N_NODES, 256), 256, 0, stream>>>(S.actF, mw1, mb1, mw2, mb2,
                                                      mw3, mb3, mw4, mb4, S.actH);
    // conv5..conv7 (decoder mid)
    for (int l = 3; l < 6; ++l) {
        run_gat_layer(S.actH, HID, Wm + (size_t)l * HID * HID,
                      ams + (size_t)l * HID, amd + (size_t)l * HID, bm + (size_t)l * HID,
                      HID, HID, HID, nullptr, S.actH, S, stream);
    }
    // conv8: 128 -> 42 (KoutPad 48), tanh result straight into d_out
    run_gat_layer(S.actH, HID, W8, a8s, a8d, b8, HID, F_IN, 48,
                  out, nullptr, S, stream);
    (void)n_in; (void)in_sizes; (void)out_size; (void)ws_size;
}
